// DLSTM_18116172054728
// MI455X (gfx1250) — compile-verified, matrix operand order checked by reading, not run on hardware
//
#include <hip/hip_runtime.h>

// Problem constants (match reference)
#define T_STEPS 512
#define B_      64
#define IN_     512
#define H_      1024
#define G4H     (4 * H_)

typedef __attribute__((ext_vector_type(16))) __bf16         v16bf;
typedef __attribute__((ext_vector_type(16))) unsigned short v16u;
typedef __attribute__((ext_vector_type(8)))  float          v8f;

#define WMMA_BF16(a, b, c) \
    __builtin_amdgcn_wmma_f32_16x16x32_bf16(false, (a), false, (b), (short)0, (c), false, false)

__device__ __forceinline__ v16bf ldfrag(const unsigned short* p) {
    v16u r = *(const v16u*)p;                // 32B contiguous -> 2x global_load_b128
    return __builtin_bit_cast(v16bf, r);
}

__device__ __forceinline__ unsigned short f2bf(float f) {
    unsigned int u = __builtin_bit_cast(unsigned int, f);
    u += 0x7fffu + ((u >> 16) & 1u);         // round-to-nearest-even
    return (unsigned short)(u >> 16);
}

__device__ __forceinline__ float sigf(float x) { return 1.0f / (1.0f + __expf(-x)); }

// ---------------------------------------------------------------------------
// fp32 -> bf16 elementwise conversion
__global__ void k_cvt_bf16(const float* __restrict__ in, unsigned short* __restrict__ out, int n) {
    int i = blockIdx.x * blockDim.x + threadIdx.x;
    if (i < n) out[i] = f2bf(in[i]);
}

// bias = b_ih + b_hh
__global__ void k_bias_sum(const float* __restrict__ a, const float* __restrict__ b,
                           float* __restrict__ o, int n) {
    int i = blockIdx.x * blockDim.x + threadIdx.x;
    if (i < n) o[i] = a[i] + b[i];
}

// ---------------------------------------------------------------------------
// Fused LSTM cell applied to one 16x16 C/D fragment set (i,f,g,o) at m-tile mt.
// C/D 16x16 f32 layout: element v -> row = mt*16 + (lane>>4)*8 + v, col = nt*16 + (lane&15)
__device__ __forceinline__ void cell_update(
    v8f gi, v8f gf, v8f gg, v8f go, int mt, int lane, int col,
    float bi, float bf, float bg, float bo,
    float* __restrict__ c, unsigned short* __restrict__ hbf_out, float* __restrict__ hf32_out)
{
    const int rbase = mt * 16 + ((lane >> 4) << 3);
    #pragma unroll
    for (int v = 0; v < 8; ++v) {
        const size_t idx = (size_t)(rbase + v) * H_ + col;
        const float iv = sigf(gi[v] + bi);
        const float fv = sigf(gf[v] + bf);
        const float gv = tanhf(gg[v] + bg);
        const float ov = sigf(go[v] + bo);
        const float cn = fv * c[idx] + iv * gv;
        const float hn = ov * tanhf(cn);
        c[idx]        = cn;
        hbf_out[idx]  = f2bf(hn);
        hf32_out[idx] = hn;
    }
}

// ---------------------------------------------------------------------------
// One LSTM layer step, fused: gates = A1 @ W1^T + A2 @ W2^T + bias, then cell.
// A1: [B_, K1] bf16 (x_t or lower-layer h), W1: [4H, K1] bf16 row-major
// A2: [B_, H_] bf16 (recurrent h),          W2: [4H, H_] bf16 row-major
// Grid: 16 blocks x 256 threads = 128 waves; wave -> (m-half 0..1, n_tile 0..63).
// Each wave computes 2 m-tiles x 4 gate tiles (8 WMMAs per k-step, 6 frag loads)
// so weights are pulled from L2 only twice per step.
__global__ __launch_bounds__(256) void k_lstm_step(
    const unsigned short* __restrict__ A1, const unsigned short* __restrict__ W1, int K1,
    const unsigned short* __restrict__ A2, const unsigned short* __restrict__ W2,
    const float* __restrict__ bias,          // [4H] = b_ih + b_hh
    float* __restrict__ c,                   // [B_, H_] in/out (elementwise in-place)
    unsigned short* __restrict__ hbf_out,    // [B_, H_] bf16 (next-step A)
    float* __restrict__ hf32_out)            // [B_, H_] fp32 (output / hT)
{
    const int wave  = (blockIdx.x * blockDim.x + threadIdx.x) >> 5;
    const int lane  = threadIdx.x & 31;
    const int nt    = wave >> 1;             // hidden tile (16 cols), 0..63
    const int mh    = wave & 1;              // batch half (32 rows)
    const int mt0   = mh * 2;                // first  batch tile (16 rows)
    const int mt1   = mh * 2 + 1;            // second batch tile (16 rows)
    const int lrow  = lane & 15;
    const int khalf = (lane >> 4) << 4;      // 0 or 16 (K sub-block per lane half)

    v8f a00 = {}, a01 = {}, a02 = {}, a03 = {};   // m-tile 0: i,f,g,o
    v8f a10 = {}, a11 = {}, a12 = {}, a13 = {};   // m-tile 1: i,f,g,o

    // ---- input contribution: A1 [B_,K1] x W1^T ----
    {
        const unsigned short* ap0 = A1 + (size_t)(mt0 * 16 + lrow) * K1 + khalf;
        const unsigned short* ap1 = A1 + (size_t)(mt1 * 16 + lrow) * K1 + khalf;
        const unsigned short* b0  = W1 + (size_t)(0 * H_ + nt * 16 + lrow) * K1 + khalf;
        const unsigned short* b1  = W1 + (size_t)(1 * H_ + nt * 16 + lrow) * K1 + khalf;
        const unsigned short* b2  = W1 + (size_t)(2 * H_ + nt * 16 + lrow) * K1 + khalf;
        const unsigned short* b3  = W1 + (size_t)(3 * H_ + nt * 16 + lrow) * K1 + khalf;
        #pragma unroll 2
        for (int k = 0; k < K1; k += 32) {
            v16bf fa0 = ldfrag(ap0 + k);
            v16bf fa1 = ldfrag(ap1 + k);
            v16bf f0  = ldfrag(b0 + k);
            v16bf f1  = ldfrag(b1 + k);
            v16bf f2  = ldfrag(b2 + k);
            v16bf f3  = ldfrag(b3 + k);
            a00 = WMMA_BF16(fa0, f0, a00);  a01 = WMMA_BF16(fa0, f1, a01);
            a02 = WMMA_BF16(fa0, f2, a02);  a03 = WMMA_BF16(fa0, f3, a03);
            a10 = WMMA_BF16(fa1, f0, a10);  a11 = WMMA_BF16(fa1, f1, a11);
            a12 = WMMA_BF16(fa1, f2, a12);  a13 = WMMA_BF16(fa1, f3, a13);
        }
    }

    // ---- recurrent contribution: A2 [B_,H_] x W2^T ----
    {
        const unsigned short* ap0 = A2 + (size_t)(mt0 * 16 + lrow) * H_ + khalf;
        const unsigned short* ap1 = A2 + (size_t)(mt1 * 16 + lrow) * H_ + khalf;
        const unsigned short* b0  = W2 + (size_t)(0 * H_ + nt * 16 + lrow) * H_ + khalf;
        const unsigned short* b1  = W2 + (size_t)(1 * H_ + nt * 16 + lrow) * H_ + khalf;
        const unsigned short* b2  = W2 + (size_t)(2 * H_ + nt * 16 + lrow) * H_ + khalf;
        const unsigned short* b3  = W2 + (size_t)(3 * H_ + nt * 16 + lrow) * H_ + khalf;
        #pragma unroll 2
        for (int k = 0; k < H_; k += 32) {
            v16bf fa0 = ldfrag(ap0 + k);
            v16bf fa1 = ldfrag(ap1 + k);
            v16bf f0  = ldfrag(b0 + k);
            v16bf f1  = ldfrag(b1 + k);
            v16bf f2  = ldfrag(b2 + k);
            v16bf f3  = ldfrag(b3 + k);
            a00 = WMMA_BF16(fa0, f0, a00);  a01 = WMMA_BF16(fa0, f1, a01);
            a02 = WMMA_BF16(fa0, f2, a02);  a03 = WMMA_BF16(fa0, f3, a03);
            a10 = WMMA_BF16(fa1, f0, a10);  a11 = WMMA_BF16(fa1, f1, a11);
            a12 = WMMA_BF16(fa1, f2, a12);  a13 = WMMA_BF16(fa1, f3, a13);
        }
    }

    // ---- fused LSTM cell on both m-tiles ----
    const int col = nt * 16 + lrow;
    const float bi = bias[0 * H_ + col];
    const float bf = bias[1 * H_ + col];
    const float bg = bias[2 * H_ + col];
    const float bo = bias[3 * H_ + col];
    cell_update(a00, a01, a02, a03, mt0, lane, col, bi, bf, bg, bo, c, hbf_out, hf32_out);
    cell_update(a10, a11, a12, a13, mt1, lane, col, bi, bf, bg, bo, c, hbf_out, hf32_out);
}

// ---------------------------------------------------------------------------
static inline int cdiv(int a, int b) { return (a + b - 1) / b; }

extern "C" void kernel_launch(void* const* d_in, const int* in_sizes, int n_in,
                              void* d_out, int out_size, void* d_ws, size_t ws_size,
                              hipStream_t stream) {
    const float* x     = (const float*)d_in[0];   // [T,B,IN]
    const float* h0    = (const float*)d_in[1];   // [L,B,H]
    const float* c0    = (const float*)d_in[2];   // [L,B,H]
    const float* W_ih0 = (const float*)d_in[3];   // [4H,IN]
    const float* W_hh0 = (const float*)d_in[4];   // [4H,H]
    const float* b_ih0 = (const float*)d_in[5];
    const float* b_hh0 = (const float*)d_in[6];
    const float* W_ih1 = (const float*)d_in[7];   // [4H,H]
    const float* W_hh1 = (const float*)d_in[8];   // [4H,H]
    const float* b_ih1 = (const float*)d_in[9];
    const float* b_hh1 = (const float*)d_in[10];
    float* out = (float*)d_out;

    // ---- workspace layout (bytes) ----
    char* ws = (char*)d_ws;
    size_t off = 0;
    auto alloc = [&](size_t bytes) { char* p = ws + off; off += (bytes + 255) & ~(size_t)255; return p; };

    unsigned short* xbf   = (unsigned short*)alloc((size_t)T_STEPS * B_ * IN_ * 2);
    unsigned short* wi0   = (unsigned short*)alloc((size_t)G4H * IN_ * 2);
    unsigned short* wh0   = (unsigned short*)alloc((size_t)G4H * H_ * 2);
    unsigned short* wi1   = (unsigned short*)alloc((size_t)G4H * H_ * 2);
    unsigned short* wh1   = (unsigned short*)alloc((size_t)G4H * H_ * 2);
    float*          bias0 = (float*)alloc((size_t)G4H * 4);
    float*          bias1 = (float*)alloc((size_t)G4H * 4);
    unsigned short* hbf0[2]; hbf0[0] = (unsigned short*)alloc((size_t)B_ * H_ * 2);
                             hbf0[1] = (unsigned short*)alloc((size_t)B_ * H_ * 2);
    unsigned short* hbf1[2]; hbf1[0] = (unsigned short*)alloc((size_t)B_ * H_ * 2);
                             hbf1[1] = (unsigned short*)alloc((size_t)B_ * H_ * 2);
    float* cws   = (float*)alloc((size_t)2 * B_ * H_ * 4);  // both layers contiguous
    float* h0f32 = (float*)alloc((size_t)B_ * H_ * 4);      // layer-0 h (fp32, for hT)

    // ---- one-time (per call) conversions ----
    {
        int n;
        n = T_STEPS * B_ * IN_; k_cvt_bf16<<<cdiv(n, 256), 256, 0, stream>>>(x, xbf, n);
        n = G4H * IN_;          k_cvt_bf16<<<cdiv(n, 256), 256, 0, stream>>>(W_ih0, wi0, n);
        n = G4H * H_;           k_cvt_bf16<<<cdiv(n, 256), 256, 0, stream>>>(W_hh0, wh0, n);
        n = G4H * H_;           k_cvt_bf16<<<cdiv(n, 256), 256, 0, stream>>>(W_ih1, wi1, n);
        n = G4H * H_;           k_cvt_bf16<<<cdiv(n, 256), 256, 0, stream>>>(W_hh1, wh1, n);
        n = B_ * H_;            k_cvt_bf16<<<cdiv(n, 256), 256, 0, stream>>>(h0,            hbf0[0], n);
        n = B_ * H_;            k_cvt_bf16<<<cdiv(n, 256), 256, 0, stream>>>(h0 + B_ * H_,  hbf1[0], n);
        n = G4H;                k_bias_sum<<<cdiv(n, 256), 256, 0, stream>>>(b_ih0, b_hh0, bias0, n);
        n = G4H;                k_bias_sum<<<cdiv(n, 256), 256, 0, stream>>>(b_ih1, b_hh1, bias1, n);
    }
    // c state: private copy (do not mutate inputs)
    hipMemcpyAsync(cws, c0, (size_t)2 * B_ * H_ * sizeof(float), hipMemcpyDeviceToDevice, stream);

    // ---- recurrent loop: 2 fused step kernels per timestep ----
    const dim3 sgrid(16), sblock(256);   // 128 waves = 2 m-halves x 64 n-tiles
    int cur = 0;
    for (int t = 0; t < T_STEPS; ++t) {
        const int nxt = cur ^ 1;
        // layer 0: A1 = x_t (K=IN), A2 = h0_prev
        k_lstm_step<<<sgrid, sblock, 0, stream>>>(
            xbf + (size_t)t * B_ * IN_, wi0, IN_,
            hbf0[cur], wh0, bias0,
            cws,                         // c layer 0
            hbf0[nxt], h0f32);
        // layer 1: A1 = h0_new (K=H), A2 = h1_prev; fp32 h -> output sequence
        k_lstm_step<<<sgrid, sblock, 0, stream>>>(
            hbf0[nxt], wi1, H_,
            hbf1[cur], wh1, bias1,
            cws + (size_t)B_ * H_,       // c layer 1
            hbf1[nxt], out + (size_t)t * B_ * H_);
        cur = nxt;
    }

    // ---- final states into d_out tail: [ops | hT(2,B,H) | cT(2,B,H)] ----
    const size_t OPS = (size_t)T_STEPS * B_ * H_;
    const size_t BH  = (size_t)B_ * H_;
    hipMemcpyAsync(out + OPS,      h0f32,                 BH * sizeof(float),
                   hipMemcpyDeviceToDevice, stream);                       // hT layer 0
    hipMemcpyAsync(out + OPS + BH, out + OPS - BH,        BH * sizeof(float),
                   hipMemcpyDeviceToDevice, stream);                       // hT layer 1 = last op
    hipMemcpyAsync(out + OPS + 2 * BH, cws,          2 * BH * sizeof(float),
                   hipMemcpyDeviceToDevice, stream);                       // cT both layers
    (void)in_sizes; (void)n_in; (void)out_size; (void)ws_size;
}
